// Block_56667798504032
// MI455X (gfx1250) — matrix operand order repaired and checked by reading, not compiled, not run on hardware
//
#include <hip/hip_runtime.h>
#include <hip/hip_bf16.h>
#include <math.h>

#define N_EMBD 2048
#define NHEAD  16
#define HDIM   128
#define BB     4
#define TT     2048
#define BT     (BB*TT)        // 8192 rows
#define QKVW   (3*N_EMBD)     // 6144

typedef __attribute__((ext_vector_type(16))) __bf16 v16bf;
typedef __attribute__((ext_vector_type(8)))  float  v8f;

union BF16Frag {
    v16bf v;
    uint4 u[2];
    unsigned short s[16];
};

__device__ __forceinline__ unsigned short f2bf(float f) {
    unsigned int x = __float_as_uint(f);
    x += 0x7fffu + ((x >> 16) & 1u);          // round-to-nearest-even
    return (unsigned short)(x >> 16);
}

// ---------------- gfx1250 async global->LDS via inline asm (ASYNCcnt-tracked).
// Existence confirmed: __has_builtin(__builtin_amdgcn_global_load_async_to_lds_b128)
// passed in the previous round; the builtin's cuda_device AS param type is not
// spellable in HIP, so use the ISA mnemonic directly.
#if defined(__has_builtin)
# if __has_builtin(__builtin_amdgcn_global_load_async_to_lds_b128)
#  define HAVE_ASYNC_LDS 1
# endif
#endif

__device__ __forceinline__ void stage_b128(const unsigned short* g, unsigned short* l) {
#ifdef HAVE_ASYNC_LDS
    unsigned int       lds_off = (unsigned int)(size_t)l;         // low 32 bits = LDS offset
    unsigned long long gaddr   = (unsigned long long)(size_t)g;   // 64-bit global VA
    asm volatile("global_load_async_to_lds_b128 %0, %1, off"
                 :: "v"(lds_off), "v"(gaddr) : "memory");
#else
    *(uint4*)l = *(const uint4*)g;
#endif
}

__device__ __forceinline__ void stage_wait() {
#ifdef HAVE_ASYNC_LDS
    asm volatile("s_wait_asynccnt 0x0" ::: "memory");
#endif
}

// ---------------------------------------------------------------- transpose+convert
// out[N,K] (bf16) = transpose(in[K,N] fp32). One-time per weight.
__global__ __launch_bounds__(256) void cvt_transpose_bf16(
        const float* __restrict__ in, unsigned short* __restrict__ out,
        int K, int N) {
    __shared__ float tile[32][33];
    int k0 = blockIdx.y * 32, n0 = blockIdx.x * 32;
    int tx = threadIdx.x & 31, ty = threadIdx.x >> 5;   // 32 x 8
#pragma unroll
    for (int i = 0; i < 32; i += 8)
        tile[ty + i][tx] = in[(size_t)(k0 + ty + i) * N + n0 + tx];
    __syncthreads();
#pragma unroll
    for (int i = 0; i < 32; i += 8)
        out[(size_t)(n0 + ty + i) * K + k0 + tx] = f2bf(tile[tx][ty + i]);
}

// ---------------------------------------------------------------- layernorm -> bf16
__global__ __launch_bounds__(256) void ln_bf16_kernel(
        const float* __restrict__ x, const float* __restrict__ w,
        const float* __restrict__ b, unsigned short* __restrict__ out) {
    __shared__ float red0[256];
    __shared__ float red1[256];
    int row = blockIdx.x;
    const float* xr = x + (size_t)row * N_EMBD;
    float vals[8];
    float s = 0.f, s2 = 0.f;
#pragma unroll
    for (int i = 0; i < 8; ++i) {
        float v = xr[threadIdx.x + i * 256];
        vals[i] = v; s += v; s2 += v * v;
    }
    red0[threadIdx.x] = s; red1[threadIdx.x] = s2;
    __syncthreads();
    for (int off = 128; off > 0; off >>= 1) {
        if (threadIdx.x < off) {
            red0[threadIdx.x] += red0[threadIdx.x + off];
            red1[threadIdx.x] += red1[threadIdx.x + off];
        }
        __syncthreads();
    }
    float mu  = red0[0] * (1.0f / N_EMBD);
    float var = red1[0] * (1.0f / N_EMBD) - mu * mu;
    float rs  = rsqrtf(var + 1e-5f);
    unsigned short* orow = out + (size_t)row * N_EMBD;
#pragma unroll
    for (int i = 0; i < 8; ++i) {
        int c = threadIdx.x + i * 256;
        orow[c] = f2bf((vals[i] - mu) * rs * w[c] + b[c]);
    }
}

// ---------------------------------------------------------------- bf16 WMMA GEMM
// C[M,N](+bias,opt GELU) = A_bf16[M,K] @ Wt_bf16[N,K]^T ; 128x128 tile, 8 waves.
#define GF_GELU    1
#define GF_BF16OUT 2

__global__ __launch_bounds__(256) void gemm_bf16_kernel(
        const unsigned short* __restrict__ A, const unsigned short* __restrict__ Wt,
        const float* __restrict__ bias, void* __restrict__ C,
        int M, int N, int K, int flags) {
    __shared__ unsigned short As[128][40];   // [m][k], stride 80B (16B-aligned rows)
    __shared__ unsigned short Bt[128][40];   // [n][k]
    int m0 = blockIdx.y * 128, n0 = blockIdx.x * 128;
    int t = threadIdx.x, lane = t & 31, wave = t >> 5;
    int wm = wave & 3, wn = wave >> 2;       // waves: 4 in M x 2 in N
    int lh = lane >> 4, ln = lane & 15;

    v8f acc[2][4];
#pragma unroll
    for (int i = 0; i < 2; ++i)
#pragma unroll
        for (int j = 0; j < 4; ++j) acc[i][j] = (v8f){0,0,0,0,0,0,0,0};

    int srow = t >> 1, scg = (t & 1) * 16;   // staging: 128 rows x 32 halves each

    const unsigned short* ag = A  + (size_t)(m0 + srow) * K + scg;
    const unsigned short* bg = Wt + (size_t)(n0 + srow) * K + scg;

    for (int k0 = 0; k0 < K; k0 += 32) {
        stage_b128(ag + k0,     &As[srow][scg]);
        stage_b128(ag + k0 + 8, &As[srow][scg + 8]);
        stage_b128(bg + k0,     &Bt[srow][scg]);
        stage_b128(bg + k0 + 8, &Bt[srow][scg + 8]);
        stage_wait();
        __syncthreads();

        BF16Frag af[2], bf[4];
#pragma unroll
        for (int i = 0; i < 2; ++i) {
            int r = wm * 32 + i * 16 + ln;
            af[i].u[0] = *(const uint4*)&As[r][lh * 8];
            af[i].u[1] = *(const uint4*)&As[r][16 + lh * 8];
        }
#pragma unroll
        for (int j = 0; j < 4; ++j) {
            int n = wn * 64 + j * 16 + ln;
            bf[j].u[0] = *(const uint4*)&Bt[n][lh * 8];
            bf[j].u[1] = *(const uint4*)&Bt[n][16 + lh * 8];
        }
#pragma unroll
        for (int i = 0; i < 2; ++i)
#pragma unroll
            for (int j = 0; j < 4; ++j)
                acc[i][j] = __builtin_amdgcn_wmma_f32_16x16x32_bf16(
                    false, af[i].v, false, bf[j].v, (short)0, acc[i][j], false, false);
        __syncthreads();
    }

#pragma unroll
    for (int i = 0; i < 2; ++i) {
#pragma unroll
        for (int j = 0; j < 4; ++j) {
            int n = n0 + wn * 64 + j * 16 + ln;
            float bi = bias[n];
#pragma unroll
            for (int v = 0; v < 8; ++v) {
                int m = m0 + wm * 32 + i * 16 + lh * 8 + v;
                float val = acc[i][j][v] + bi;
                if (flags & GF_GELU)
                    val = 0.5f * val * (1.0f + erff(val * 0.70710678118654752f));
                if (flags & GF_BF16OUT)
                    ((unsigned short*)C)[(size_t)m * N + n] = f2bf(val);
                else
                    ((float*)C)[(size_t)m * N + n] = val;
            }
        }
    }
}

// ---------------------------------------------------------------- causal flash attention
// qkv bf16 [BT, 6144]; per (b,h): S^T = K.Q^T (WMMA), online softmax per query
// (lane-pair reduction), O^T = V^T.P^T (WMMA). 4 waves/block share K/V staging.
__global__ __launch_bounds__(128) void attn_kernel(
        const unsigned short* __restrict__ qkv, float* __restrict__ y) {
    __shared__ unsigned short Ks[32][136];   // stride 272B: 16B-aligned, conflict-free cols
    __shared__ unsigned short Vs[32][136];
    int t = threadIdx.x, lane = t & 31, wave = t >> 5;
    int grp = blockIdx.x & 31;               // 32 query-groups of 64
    int bh  = blockIdx.x >> 5;               // 0..63
    int h = bh & 15, b = bh >> 4;
    int q0 = (grp * 4 + wave) * 16;
    int lh = lane >> 4, ln = lane & 15;
    const float scale = 0.08838834764831845f;   // 1/sqrt(128)

    const unsigned short* base = qkv + (size_t)b * TT * QKVW + (size_t)h * HDIM;

    // Preload Q fragments (B-operand of S^T): 4 d-chunks of 32
    BF16Frag qf[4];
    {
        const unsigned short* qrow = base + (size_t)(q0 + ln) * QKVW;
#pragma unroll
        for (int dc = 0; dc < 4; ++dc) {
            qf[dc].u[0] = *(const uint4*)(qrow + dc * 32 + lh * 8);
            qf[dc].u[1] = *(const uint4*)(qrow + dc * 32 + 16 + lh * 8);
        }
    }

    v8f oacc[8];
#pragma unroll
    for (int dt = 0; dt < 8; ++dt) oacc[dt] = (v8f){0,0,0,0,0,0,0,0};
    float mrun = -1e30f, lrun = 0.f;

    int nkb = grp * 2 + 2;                   // key blocks of 32 (causal)
    for (int kb = 0; kb < nkb; ++kb) {
        {   // cooperative K/V staging: 128 threads, 32 rows x 128 halves each
            int r = t >> 2, cg = (t & 3) * 32;
            const unsigned short* src = base + (size_t)(kb * 32 + r) * QKVW;
#pragma unroll
            for (int i = 0; i < 4; ++i) {
                stage_b128(src + N_EMBD     + cg + i * 8, &Ks[r][cg + i * 8]);
                stage_b128(src + 2 * N_EMBD + cg + i * 8, &Vs[r][cg + i * 8]);
            }
            stage_wait();
        }
        __syncthreads();

        // S^T tiles: rows = keys (2 tiles of 16), cols = queries
        v8f st[2];
        st[0] = (v8f){0,0,0,0,0,0,0,0};
        st[1] = (v8f){0,0,0,0,0,0,0,0};
#pragma unroll
        for (int mt = 0; mt < 2; ++mt) {
            int r = mt * 16 + ln;
#pragma unroll
            for (int dc = 0; dc < 4; ++dc) {
                BF16Frag af;
                af.u[0] = *(const uint4*)&Ks[r][dc * 32 + lh * 8];
                af.u[1] = *(const uint4*)&Ks[r][dc * 32 + 16 + lh * 8];
                st[mt] = __builtin_amdgcn_wmma_f32_16x16x32_bf16(
                    false, af.v, false, qf[dc].v, (short)0, st[mt], false, false);
            }
        }

        // scale + causal mask; online softmax (lane L pairs with L^16: same query)
        int q = q0 + ln;
        float e0[8], e1[8];
        float lmax = -1e30f;
#pragma unroll
        for (int v = 0; v < 8; ++v) {
            int k0i = kb * 32 + lh * 8 + v;
            float s0 = (k0i      <= q) ? st[0][v] * scale : -1e30f;
            float s1 = (k0i + 16 <= q) ? st[1][v] * scale : -1e30f;
            e0[v] = s0; e1[v] = s1;
            lmax = fmaxf(lmax, fmaxf(s0, s1));
        }
        lmax = fmaxf(lmax, __shfl_xor(lmax, 16, 32));
        float mnew = fmaxf(mrun, lmax);
        float corr = expf(mrun - mnew);
        float lsum = 0.f;
#pragma unroll
        for (int v = 0; v < 8; ++v) {
            e0[v] = expf(e0[v] - mnew);
            e1[v] = expf(e1[v] - mnew);
            lsum += e0[v] + e1[v];
        }
        lsum += __shfl_xor(lsum, 16, 32);
        lrun = lrun * corr + lsum;
        mrun = mnew;
#pragma unroll
        for (int dt = 0; dt < 8; ++dt) oacc[dt] *= corr;

        // Build P^T B-operand: one cross-half shuffle assembles 16 keys per lane
        BF16Frag pf;
#pragma unroll
        for (int v = 0; v < 8; ++v) {
            float o0 = __shfl_xor(e0[v], 16, 32);
            float o1 = __shfl_xor(e1[v], 16, 32);
            float lo = lh ? o1 : e0[v];
            float hi = lh ? e1[v] : o0;
            pf.s[v]     = f2bf(lo);
            pf.s[v + 8] = f2bf(hi);
        }

        // O^T += V^T @ P^T : 8 d-tiles of 16
#pragma unroll
        for (int dt = 0; dt < 8; ++dt) {
            BF16Frag af;
            int d = dt * 16 + ln;
#pragma unroll
            for (int g = 0; g < 8; ++g) {
                af.s[g]     = Vs[lh * 16 + g][d];
                af.s[8 + g] = Vs[lh * 16 + 8 + g][d];
            }
            oacc[dt] = __builtin_amdgcn_wmma_f32_16x16x32_bf16(
                false, af.v, false, pf.v, (short)0, oacc[dt], false, false);
        }
        __syncthreads();
    }

    float inv = 1.0f / lrun;
    float* yout = y + (size_t)(b * TT + q0 + ln) * N_EMBD + h * HDIM;
#pragma unroll
    for (int dt = 0; dt < 8; ++dt)
#pragma unroll
        for (int v = 0; v < 8; ++v)
            yout[dt * 16 + lh * 8 + v] = oacc[dt][v] * inv;
}

// ---------------------------------------------------------------- launch
extern "C" void kernel_launch(void* const* d_in, const int* in_sizes, int n_in,
                              void* d_out, int out_size, void* d_ws, size_t ws_size,
                              hipStream_t stream) {
    const float* x      = (const float*)d_in[0];
    const float* ln1_w  = (const float*)d_in[1];
    const float* ln1_b  = (const float*)d_in[2];
    const float* attn_w = (const float*)d_in[3];
    const float* attn_b = (const float*)d_in[4];
    const float* ln2_w  = (const float*)d_in[5];
    const float* ln2_b  = (const float*)d_in[6];
    const float* fc_w   = (const float*)d_in[7];
    const float* fc_b   = (const float*)d_in[8];
    const float* proj_w = (const float*)d_in[9];
    const float* proj_b = (const float*)d_in[10];

    char* ws = (char*)d_ws;
    size_t off = 0;
    auto alloc = [&](size_t bytes) {
        void* p = ws + off;
        off = (off + bytes + 255) & ~(size_t)255;
        return p;
    };
    unsigned short* attn_wt = (unsigned short*)alloc((size_t)N_EMBD * QKVW * 2);       // [6144,2048]
    unsigned short* fc_wt   = (unsigned short*)alloc((size_t)N_EMBD * 4 * N_EMBD * 2); // [8192,2048]
    unsigned short* proj_wt = (unsigned short*)alloc((size_t)4 * N_EMBD * N_EMBD * 2); // [2048,8192]
    unsigned short* h1      = (unsigned short*)alloc((size_t)BT * N_EMBD * 2);
    unsigned short* qkvb    = (unsigned short*)alloc((size_t)BT * QKVW * 2);
    float*          ybuf    = (float*)alloc((size_t)BT * N_EMBD * 4);
    unsigned short* h2      = (unsigned short*)alloc((size_t)BT * N_EMBD * 2);
    unsigned short* gbuf    = (unsigned short*)alloc((size_t)BT * 4 * N_EMBD * 2);

    // one-time transposed bf16 weights
    cvt_transpose_bf16<<<dim3(QKVW / 32, N_EMBD / 32), 256, 0, stream>>>(
        attn_w, attn_wt, N_EMBD, QKVW);
    cvt_transpose_bf16<<<dim3(4 * N_EMBD / 32, N_EMBD / 32), 256, 0, stream>>>(
        fc_w, fc_wt, N_EMBD, 4 * N_EMBD);
    cvt_transpose_bf16<<<dim3(N_EMBD / 32, 4 * N_EMBD / 32), 256, 0, stream>>>(
        proj_w, proj_wt, 4 * N_EMBD, N_EMBD);

    ln_bf16_kernel<<<BT, 256, 0, stream>>>(x, ln1_w, ln1_b, h1);

    dim3 g1(QKVW / 128, BT / 128);
    gemm_bf16_kernel<<<g1, 256, 0, stream>>>(h1, attn_wt, attn_b, qkvb,
                                             BT, QKVW, N_EMBD, GF_BF16OUT);

    attn_kernel<<<BB * NHEAD * (TT / 64), 128, 0, stream>>>(qkvb, ybuf);

    ln_bf16_kernel<<<BT, 256, 0, stream>>>(ybuf, ln2_w, ln2_b, h2);

    dim3 g2(4 * N_EMBD / 128, BT / 128);
    gemm_bf16_kernel<<<g2, 256, 0, stream>>>(h2, fc_wt, fc_b, gbuf,
                                             BT, 4 * N_EMBD, N_EMBD,
                                             GF_GELU | GF_BF16OUT);

    dim3 g3(N_EMBD / 128, BT / 128);
    gemm_bf16_kernel<<<g3, 256, 0, stream>>>(gbuf, proj_wt, proj_b, d_out,
                                             BT, N_EMBD, 4 * N_EMBD, 0);
}
